// CrossAttention_3985729651422
// MI455X (gfx1250) — matrix operand order repaired and checked by reading, not compiled, not run on hardware
//
#include <hip/hip_runtime.h>
#include <hip/hip_bf16.h>

typedef _Float16 half_t;
typedef __attribute__((ext_vector_type(16))) _Float16 v16h;
typedef __attribute__((ext_vector_type(8)))  _Float16 v8h;
typedef __attribute__((ext_vector_type(4)))  _Float16 v4h;
typedef __attribute__((ext_vector_type(8)))  float    v8f;
typedef __attribute__((ext_vector_type(4)))  float    v4f;

#define DEV __device__ __forceinline__

// ---- WMMA fragment helpers (layouts per CDNA5 ISA 7.12.2, wave32) ----

// A 16x32 f16: lanes 0-15 row m=lane, K = {hi? 8..15 : 0..7} in v0-3, {hi? 24..31 : 16..23} in v4-7.
DEV v16h frag_a(const half_t* p_lo, const half_t* p_hi) {
  union { v16h v; v8h h[2]; } u;
  u.h[0] = *(const v8h*)p_lo;   // 8 halves: K base .. base+7
  u.h[1] = *(const v8h*)p_hi;   // 8 halves: K base+16 .. base+23
  return u.v;
}
// B 32x16 f16 from an [n][k] row-major buffer: lane n=lane&15 reads
// K = hi*8 + 0..7 and K = 16 + hi*8 + 0..7.
DEV v16h frag_b(const half_t* p) {
  union { v16h v; v8h h[2]; } u;
  u.h[0] = *(const v8h*)p;
  u.h[1] = *(const v8h*)(p + 16);
  return u.v;
}
DEV v8f wmma_f16(v16h a, v16h b, v8f c) {
  return __builtin_amdgcn_wmma_f32_16x16x32_f16(false, a, false, b, (short)0, c,
                                                false, false);
}
DEV v8h cvt8(v8f a) {
  v8h r;
#pragma unroll
  for (int i = 0; i < 8; ++i) r[i] = (_Float16)a[i];
  return r;
}

// ---- Generic WMMA GEMM: C[M x N] = A[M x K] * Bt^T  (Bt stored [N][K] f16) ----
// block = 128 threads (4 waves); block tile 128(M) x 64(N); k-step 32.
template <bool A_F32, bool OUT_F16, bool BIAS>
__global__ __launch_bounds__(128) void gemm_wmma_k(
    const void* __restrict__ Ap, const half_t* __restrict__ Bt,
    void* __restrict__ Cp, const float* __restrict__ bias,
    int Mload, int K, int lda, int ldc, float scale) {
  constexpr int LA = 40, LB = 40, LC = 68;  // padded strides (bank-conflict free)
  __shared__ union {
    struct { half_t As[128 * LA]; half_t Bs[64 * LB]; } s;
    float Cs[128 * LC];
  } u;

  const int t = threadIdx.x;
  const int w = t >> 5, lane = t & 31, ln = lane & 15, hi = lane >> 4;
  const int m0 = blockIdx.x * 128, n0 = blockIdx.y * 64;

  v8f acc[2][4] = {};

  for (int k0 = 0; k0 < K; k0 += 32) {
    __syncthreads();
    // ---- stage A tile 128x32 (convert f32->f16 on the fly if needed) ----
    if constexpr (A_F32) {
      const float* A = (const float*)Ap;
#pragma unroll
      for (int j = 0; j < 8; ++j) {
        int chunk = j * 128 + t;
        int r = chunk >> 3, c = (chunk & 7) * 4;
        int rg = m0 + r;
        v4f d = {};
        if (rg < Mload) {
          d = *(const v4f*)(A + (long)rg * lda + k0 + c);
          if (k0 + 32 < K) __builtin_prefetch(A + (long)rg * lda + k0 + 32 + c, 0, 1);
        }
        v4h hd;
        hd[0] = (_Float16)d[0]; hd[1] = (_Float16)d[1];
        hd[2] = (_Float16)d[2]; hd[3] = (_Float16)d[3];
        *(v4h*)&u.s.As[r * LA + c] = hd;
      }
    } else {
      const half_t* A = (const half_t*)Ap;
#pragma unroll
      for (int j = 0; j < 4; ++j) {
        int chunk = j * 128 + t;
        int r = chunk >> 2, c = (chunk & 3) * 8;
        int rg = m0 + r;
        v8h d = {};
        if (rg < Mload) {
          d = *(const v8h*)(A + (long)rg * lda + k0 + c);
          if (k0 + 32 < K) __builtin_prefetch(A + (long)rg * lda + k0 + 32 + c, 0, 1);
        }
        *(v8h*)&u.s.As[r * LA + c] = d;
      }
    }
    // ---- stage B tile 64x32 from Bt [N][K] ----
#pragma unroll
    for (int j = 0; j < 2; ++j) {
      int chunk = j * 128 + t;
      int r = chunk >> 2, c = (chunk & 3) * 8;
      *(v8h*)&u.s.Bs[r * LB + c] = *(const v8h*)(Bt + (long)(n0 + r) * K + k0 + c);
    }
    __syncthreads();

    v16h af[2], bf[4];
#pragma unroll
    for (int mt = 0; mt < 2; ++mt) {
      const half_t* p = &u.s.As[(w * 32 + mt * 16 + ln) * LA + hi * 8];
      af[mt] = frag_a(p, p + 16);
    }
#pragma unroll
    for (int nt = 0; nt < 4; ++nt)
      bf[nt] = frag_b(&u.s.Bs[(nt * 16 + ln) * LB + hi * 8]);
#pragma unroll
    for (int mt = 0; mt < 2; ++mt)
#pragma unroll
      for (int nt = 0; nt < 4; ++nt)
        acc[mt][nt] = wmma_f16(af[mt], bf[nt], acc[mt][nt]);
  }

  // ---- epilogue: stage f32 in LDS, then coalesced global stores ----
  __syncthreads();
#pragma unroll
  for (int nt = 0; nt < 4; ++nt) {
    float bv = BIAS ? bias[n0 + nt * 16 + ln] : 0.f;
#pragma unroll
    for (int mt = 0; mt < 2; ++mt)
#pragma unroll
      for (int v = 0; v < 8; ++v) {
        int r = w * 32 + mt * 16 + hi * 8 + v;
        u.Cs[r * LC + nt * 16 + ln] = acc[mt][nt][v] * scale + bv;
      }
  }
  __syncthreads();
  if constexpr (OUT_F16) {
    half_t* C = (half_t*)Cp;
#pragma unroll
    for (int j = 0; j < 8; ++j) {
      int chunk = j * 128 + t;
      int r = chunk >> 3, c = (chunk & 7) * 8;
      const float* src = &u.Cs[r * LC + c];
      v8h d;
#pragma unroll
      for (int i = 0; i < 8; ++i) d[i] = (_Float16)src[i];
      *(v8h*)(C + (long)(m0 + r) * ldc + n0 + c) = d;
    }
  } else {
    float* C = (float*)Cp;
#pragma unroll
    for (int j = 0; j < 16; ++j) {
      int chunk = j * 128 + t;
      int r = chunk >> 4, c = (chunk & 15) * 4;
      *(v4f*)(C + (long)(m0 + r) * ldc + n0 + c) = *(const v4f*)&u.Cs[r * LC + c];
    }
  }
}

// ---- Fused attention: per (head, 128-row S tile) ----
// scores^T = K(96x64) * Q^T(64x128); softmax over l (rows); out^T = V^T(64x96)*P^T(96x128)
__global__ __launch_bounds__(128) void attn_k(
    const half_t* __restrict__ Qh, const half_t* __restrict__ Kh,
    const half_t* __restrict__ Vt, half_t* __restrict__ O) {
  constexpr int LQ = 72, LK = 72, LV = 104, LP = 104;
  __shared__ union {
    struct { half_t Qs[128 * LQ]; half_t Ks[96 * LK]; } a;  // score phase
    half_t Ps[128 * LP];                                    // P stored [s][l]
  } u;
  __shared__ half_t Vs[64 * LV];  // V^T [d][l], l padded to 96

  const int t = threadIdx.x;
  const int w = t >> 5, lane = t & 31, ln = lane & 15, hi = lane >> 4;
  const int s0 = blockIdx.x * 128;
  const int bh = blockIdx.y, b = bh >> 3, h = bh & 7;
  const long qbase = ((long)(b * 4096 + s0)) * 512 + h * 64;
  const long kbase = ((long)(b * 77)) * 512 + h * 64;  // rows up to 634 < 640 pad
  const long vbase = (long)bh * 64 * 96;

  // stage Q (128x64), K (96x64) as [row][d], V^T (64x96) as [d][l]
#pragma unroll
  for (int j = 0; j < 8; ++j) {
    int chunk = j * 128 + t, r = chunk >> 3, c = (chunk & 7) * 8;
    *(v8h*)&u.a.Qs[r * LQ + c] = *(const v8h*)(Qh + qbase + (long)r * 512 + c);
  }
#pragma unroll
  for (int j = 0; j < 6; ++j) {
    int chunk = j * 128 + t, r = chunk >> 3, c = (chunk & 7) * 8;
    *(v8h*)&u.a.Ks[r * LK + c] = *(const v8h*)(Kh + kbase + (long)r * 512 + c);
  }
#pragma unroll
  for (int j = 0; j < 6; ++j) {
    int chunk = j * 128 + t, r = chunk / 12, c = (chunk % 12) * 8;
    *(v8h*)&Vs[r * LV + c] = *(const v8h*)(Vt + vbase + (long)r * 96 + c);
  }
  __syncthreads();

  // scores^T: M = l (5 tiles, l<80), N = s (wave w owns n-tiles {2w,2w+1})
  v8f sc[5][2] = {};
#pragma unroll
  for (int ks = 0; ks < 2; ++ks) {
    v16h qb[2];
#pragma unroll
    for (int n2 = 0; n2 < 2; ++n2)
      qb[n2] = frag_b(&u.a.Qs[((w * 2 + n2) * 16 + ln) * LQ + ks * 32 + hi * 8]);
#pragma unroll
    for (int mt = 0; mt < 5; ++mt) {
      const half_t* p = &u.a.Ks[(mt * 16 + ln) * LK + ks * 32 + hi * 8];
      v16h af = frag_a(p, p + 16);
#pragma unroll
      for (int n2 = 0; n2 < 2; ++n2) sc[mt][n2] = wmma_f16(af, qb[n2], sc[mt][n2]);
    }
  }
  __syncthreads();  // Qs/Ks dead; region reused as Ps

  // softmax over l: in-lane over 5x8 values + one xor-16 exchange
#pragma unroll
  for (int n2 = 0; n2 < 2; ++n2) {
    float mx = -3.0e38f;
#pragma unroll
    for (int mt = 0; mt < 5; ++mt)
#pragma unroll
      for (int v = 0; v < 8; ++v) {
        int l = mt * 16 + hi * 8 + v;
        if (l < 77) mx = fmaxf(mx, sc[mt][n2][v]);
      }
    mx = fmaxf(mx, __shfl_xor(mx, 16, 32));
    float sum = 0.f;
#pragma unroll
    for (int mt = 0; mt < 5; ++mt)
#pragma unroll
      for (int v = 0; v < 8; ++v) {
        int l = mt * 16 + hi * 8 + v;
        float p = (l < 77) ? __expf(sc[mt][n2][v] - mx) : 0.f;
        sc[mt][n2][v] = p;
        sum += p;
      }
    sum += __shfl_xor(sum, 16, 32);
    float inv = 1.f / sum;
    int srow = (w * 2 + n2) * 16 + ln;
#pragma unroll
    for (int mt = 0; mt < 5; ++mt) {  // 8 consecutive l per lane -> one b128 store
      v8h pk;
#pragma unroll
      for (int v = 0; v < 8; ++v) pk[v] = (_Float16)(sc[mt][n2][v] * inv);
      *(v8h*)&u.Ps[srow * LP + mt * 16 + hi * 8] = pk;
    }
  }
  {  // zero-pad P columns l = 80..95
    int srow = w * 32 + lane;
    v8h z = {};
    *(v8h*)&u.Ps[srow * LP + 80] = z;
    *(v8h*)&u.Ps[srow * LP + 88] = z;
  }
  __syncthreads();

  // out^T = V^T * P^T : M = d (4 tiles), N = s (same 2 n-tiles), K = 96
  v8f o[4][2] = {};
#pragma unroll
  for (int ks = 0; ks < 3; ++ks) {
    v16h pb[2];
#pragma unroll
    for (int n2 = 0; n2 < 2; ++n2)
      pb[n2] = frag_b(&u.Ps[((w * 2 + n2) * 16 + ln) * LP + ks * 32 + hi * 8]);
#pragma unroll
    for (int mt = 0; mt < 4; ++mt) {
      const half_t* p = &Vs[(mt * 16 + ln) * LV + ks * 32 + hi * 8];
      v16h af = frag_a(p, p + 16);
#pragma unroll
      for (int n2 = 0; n2 < 2; ++n2) o[mt][n2] = wmma_f16(af, pb[n2], o[mt][n2]);
    }
  }
  // store attn_out rows: lane holds fixed s, 8 consecutive d -> b128 f16 stores
#pragma unroll
  for (int n2 = 0; n2 < 2; ++n2) {
    long row = (long)(b * 4096 + s0 + (w * 2 + n2) * 16 + ln);
#pragma unroll
    for (int mt = 0; mt < 4; ++mt)
      *(v8h*)(O + row * 512 + h * 64 + mt * 16 + hi * 8) = cvt8(o[mt][n2]);
  }
}

// ---- prep kernels ----
__global__ __launch_bounds__(256) void transpose_w_k(const float* __restrict__ W,
                                                     half_t* __restrict__ Wt,
                                                     int K, int N) {
  int t = blockIdx.x * 256 + threadIdx.x;
  if (t >= K * N) return;
  int n = t / K, k = t - n * K;
  Wt[t] = (_Float16)W[(long)k * N + n];  // Wt[n][k]
}

__global__ __launch_bounds__(256) void prep_vt_k(const half_t* __restrict__ Vh,
                                                 half_t* __restrict__ Vt) {
  int t = blockIdx.x * 256 + threadIdx.x;  // 64*64*96
  int l = t % 96, rest = t / 96, d = rest & 63, bh = rest >> 6;
  int b = bh >> 3, h = bh & 7;
  half_t v = (_Float16)0.f;
  if (l < 77) v = Vh[(long)(b * 77 + l) * 512 + h * 64 + d];
  Vt[t] = v;  // [bh][d][l], zero-padded to l=96
}

extern "C" void kernel_launch(void* const* d_in, const int* in_sizes, int n_in,
                              void* d_out, int out_size, void* d_ws, size_t ws_size,
                              hipStream_t stream) {
  const float* x   = (const float*)d_in[0];
  const float* ctx = (const float*)d_in[1];
  const float* Wq  = (const float*)d_in[2];
  const float* Wk  = (const float*)d_in[3];
  const float* Wv  = (const float*)d_in[4];
  const float* Wo  = (const float*)d_in[5];
  const float* bo  = (const float*)d_in[6];
  float* out = (float*)d_out;

  half_t* ws = (half_t*)d_ws;
  size_t off = 0;
  half_t* Qh  = ws + off; off += (size_t)32768 * 512;
  half_t* AO  = ws + off; off += (size_t)32768 * 512;
  half_t* Kh  = ws + off; off += (size_t)640 * 512;
  half_t* Vh  = ws + off; off += (size_t)640 * 512;
  half_t* Vt  = ws + off; off += (size_t)64 * 64 * 96;
  half_t* WqT = ws + off; off += (size_t)512 * 1024;
  half_t* WkT = ws + off; off += (size_t)512 * 768;
  half_t* WvT = ws + off; off += (size_t)512 * 768;
  half_t* WoT = ws + off; off += (size_t)1024 * 512;

  transpose_w_k<<<(1024 * 512) / 256, 256, 0, stream>>>(Wq, WqT, 1024, 512);
  transpose_w_k<<<(768 * 512) / 256, 256, 0, stream>>>(Wk, WkT, 768, 512);
  transpose_w_k<<<(768 * 512) / 256, 256, 0, stream>>>(Wv, WvT, 768, 512);
  transpose_w_k<<<(512 * 1024) / 256, 256, 0, stream>>>(Wo, WoT, 512, 1024);

  // Q = x @ Wq, scale 1/sqrt(64) folded in
  gemm_wmma_k<true, true, false><<<dim3(256, 8), 128, 0, stream>>>(
      x, WqT, Qh, nullptr, 32768, 1024, 1024, 512, 0.125f);
  // K/V = ctx @ Wk/Wv (M padded 616 -> 640)
  gemm_wmma_k<true, true, false><<<dim3(5, 8), 128, 0, stream>>>(
      ctx, WkT, Kh, nullptr, 616, 768, 768, 512, 1.0f);
  gemm_wmma_k<true, true, false><<<dim3(5, 8), 128, 0, stream>>>(
      ctx, WvT, Vh, nullptr, 616, 768, 768, 512, 1.0f);
  prep_vt_k<<<(64 * 64 * 96) / 256, 256, 0, stream>>>(Vh, Vt);
  attn_k<<<dim3(32, 64), 128, 0, stream>>>(Qh, Kh, Vt, AO);
  // out = AO @ Wo + bo  (f32 output)
  gemm_wmma_k<false, false, true><<<dim3(256, 16), 128, 0, stream>>>(
      AO, WoT, out, bo, 32768, 512, 512, 1024, 1.0f);
}
